// BaseAttnPredictNet_71021579206948
// MI455X (gfx1250) — matrix operand order repaired and checked by reading, not compiled, not run on hardware
//
#include <hip/hip_runtime.h>
#include <hip/hip_bf16.h>
#include <cstdint>

#define DMODEL 512
#define HEADS  8
#define DH     64
#define QLEN   2048
#define KLEN   2048
#define KPAD   2112      // KLEN + 1 zero-attn slot, padded to multiple of 64
#define BATCH  2
#define ATT_SCALE 0.125f
#define NEG_INF  -1.0e30f

typedef __attribute__((ext_vector_type(16))) __bf16 v16bf;
typedef __attribute__((ext_vector_type(8)))  float  v8f;

union BFrag { v16bf v; uint4 q[2]; unsigned short s[16]; };

__device__ __forceinline__ unsigned short f32_to_bf16(float f) {
  union { float f; unsigned u; } x; x.f = f;
  unsigned r = 0x7fffu + ((x.u >> 16) & 1u);
  return (unsigned short)((x.u + r) >> 16);
}

// A-operand fragment (16 x 32 tile): lane holds row M = lane&15,
// K chunks {0..7,16..23} (lanes 0-15) / {8..15,24..31} (lanes 16-31).
__device__ __forceinline__ void load_fragA(BFrag& f, const unsigned short* row, int hi) {
  f.q[0] = *reinterpret_cast<const uint4*>(row + (hi ? 8 : 0));
  f.q[1] = *reinterpret_cast<const uint4*>(row + 16 + (hi ? 8 : 0));
}
// B-operand fragment from B^T storage (row = output column N = lane&15):
// K chunks {0..7,16..23} (lanes 0-15) / {8..15,24..31} (lanes 16-31).
__device__ __forceinline__ void load_fragB(BFrag& f, const unsigned short* row, int hi) {
  f.q[0] = *reinterpret_cast<const uint4*>(row + hi * 8);
  f.q[1] = *reinterpret_cast<const uint4*>(row + 16 + hi * 8);
}

__device__ __forceinline__ v8f wmma_bf16(const BFrag& a, const BFrag& b, v8f c) {
  return __builtin_amdgcn_wmma_f32_16x16x32_bf16(false, a.v, false, b.v,
                                                 (short)0, c, false, false);
}

__device__ __forceinline__ v8f v8f_zero() {
  v8f z;
#pragma unroll
  for (int i = 0; i < 8; i++) z[i] = 0.f;
  return z;
}

// ---- CDNA5 async global->LDS copy (ASYNCcnt-tracked, bypasses VGPRs) ----
__device__ __forceinline__ void async_ld128(void* lds, const void* g) {
  unsigned ldsOff = (unsigned)(uintptr_t)lds;               // low 32 bits of generic
  unsigned long long ga = (unsigned long long)(uintptr_t)g; // LDS ptr = LDS offset
  asm volatile("global_load_async_to_lds_b128 %0, %1, off"
               :: "v"(ldsOff), "v"(ga) : "memory");
}
__device__ __forceinline__ void wait_async() {
  asm volatile("s_wait_asynccnt 0x0" ::: "memory");
}

// ---------------- LayerNorm: one wave per 512-elem row ----------------
__global__ __launch_bounds__(256) void ln_q_kernel(
    const float* __restrict__ x, const float* __restrict__ gamma,
    const float* __restrict__ beta, unsigned short* __restrict__ out,
    unsigned short* __restrict__ raw) {
  int lane = threadIdx.x & 31, w = threadIdx.x >> 5;
  int row = blockIdx.x * 8 + w;
  const float* xr = x + (size_t)row * DMODEL;
  float v[16], s = 0.f, s2 = 0.f;
#pragma unroll
  for (int t = 0; t < 16; t++) {
    float f = xr[lane + t * 32];
    v[t] = f; s += f; s2 += f * f;
  }
#pragma unroll
  for (int m = 16; m > 0; m >>= 1) { s += __shfl_xor(s, m, 32); s2 += __shfl_xor(s2, m, 32); }
  float mean = s * (1.f / DMODEL);
  float var  = s2 * (1.f / DMODEL) - mean * mean;
  float rstd = rsqrtf(var + 1e-5f);
#pragma unroll
  for (int t = 0; t < 16; t++) {
    int i = lane + t * 32;
    out[(size_t)row * DMODEL + i] = f32_to_bf16((v[t] - mean) * rstd * gamma[i] + beta[i]);
    raw[(size_t)row * DMODEL + i] = f32_to_bf16(v[t]);
  }
}

// LN over K/V with zero-attn slot (row KLEN -> LN(0)=beta) and zero padding.
__global__ __launch_bounds__(256) void ln_kv_kernel(
    const float* __restrict__ x, const float* __restrict__ gamma,
    const float* __restrict__ beta, unsigned short* __restrict__ out) {
  int lane = threadIdx.x & 31, w = threadIdx.x >> 5;
  int row = blockIdx.x * 8 + w;
  int b = row / KPAD, j = row % KPAD;
  if (j >= KLEN) {
#pragma unroll
    for (int t = 0; t < 16; t++) {
      int i = lane + t * 32;
      out[(size_t)row * DMODEL + i] = (j == KLEN) ? f32_to_bf16(beta[i]) : (unsigned short)0;
    }
    return;
  }
  const float* xr = x + (size_t)(b * KLEN + j) * DMODEL;
  float v[16], s = 0.f, s2 = 0.f;
#pragma unroll
  for (int t = 0; t < 16; t++) {
    float f = xr[lane + t * 32];
    v[t] = f; s += f; s2 += f * f;
  }
#pragma unroll
  for (int m = 16; m > 0; m >>= 1) { s += __shfl_xor(s, m, 32); s2 += __shfl_xor(s2, m, 32); }
  float mean = s * (1.f / DMODEL);
  float var  = s2 * (1.f / DMODEL) - mean * mean;
  float rstd = rsqrtf(var + 1e-5f);
#pragma unroll
  for (int t = 0; t < 16; t++) {
    int i = lane + t * 32;
    out[(size_t)row * DMODEL + i] = f32_to_bf16((v[t] - mean) * rstd * gamma[i] + beta[i]);
  }
}

// ---- weight transpose + scale + f32->bf16: Wt[n][k] = bf16(W[k][n]*scale) ----
__global__ __launch_bounds__(256) void wtrans_kernel(
    const float* __restrict__ W, unsigned short* __restrict__ Wt, int K, int N,
    float scale) {
  __shared__ float tile[32][33];
  int ntiles = N / 32;
  int bx = blockIdx.x % ntiles, by = blockIdx.x / ntiles;
  int k0 = by * 32, n0 = bx * 32;
  int col = threadIdx.x & 31, rg = threadIdx.x >> 5;
#pragma unroll
  for (int i = 0; i < 4; i++) {
    int r = rg * 4 + i;
    tile[r][col] = W[(size_t)(k0 + r) * N + n0 + col];
  }
  __syncthreads();
#pragma unroll
  for (int i = 0; i < 4; i++) {
    int r = rg * 4 + i;   // output row (n index)
    Wt[(size_t)(n0 + r) * K + k0 + col] = f32_to_bf16(tile[col][r] * scale);
  }
}

// -------- vh [B][KPAD][D] bf16 -> vt [B][H][DH][KPAD] bf16 --------
__global__ __launch_bounds__(256) void vtrans_kernel(
    const unsigned short* __restrict__ vh, unsigned short* __restrict__ vt) {
  __shared__ unsigned short t[64][72];
  int blk = blockIdx.x;
  int jt = blk % (KPAD / 64);
  int h  = (blk / (KPAD / 64)) % HEADS;
  int b  = blk / ((KPAD / 64) * HEADS);
  int j0 = jt * 64;
  int tid = threadIdx.x;
  for (int c = tid; c < 512; c += 256) {
    int jj = c >> 3, col = (c & 7) * 8;
    *(uint4*)&t[jj][col] =
        *(const uint4*)(vh + (size_t)(b * KPAD + j0 + jj) * DMODEL + h * DH + col);
  }
  __syncthreads();
  for (int c = tid; c < 512; c += 256) {
    int d = c >> 3, j8 = (c & 7) * 8;
    union { uint4 u; unsigned short s[8]; } o;
#pragma unroll
    for (int e = 0; e < 8; e++) o.s[e] = t[j8 + e][d];
    *(uint4*)(vt + ((size_t)(b * HEADS + h) * DH + d) * KPAD + j0 + j8) = o.u;
  }
}

// ---------------- generic bf16 GEMM: C = A * B^T (+bias) (+acc) ----------------
// OUT: 0 = f32 out, 1 = bf16 out, 2 = both.
template <int OUT, bool ACC, bool BIAS>
__global__ __launch_bounds__(256) void gemm_bf16_kernel(
    const unsigned short* __restrict__ A, int lda,
    const unsigned short* __restrict__ Bt, int ldb,
    float* __restrict__ Cf, unsigned short* __restrict__ Cb, int ldc,
    int M, int N, int Kd, const float* __restrict__ bias) {
  __shared__ unsigned short As[2][128][40];
  __shared__ unsigned short Bs[2][128][40];
  int tid = threadIdx.x, lane = tid & 31, w = tid >> 5;
  int hi = lane >> 4, ln = lane & 15;
  int wm = w & 1, wn = w >> 1;        // 2 x 4 wave grid; wave tile = 64M x 32N
  int nbn = N / 128;
  int m0 = (blockIdx.x / nbn) * 128, n0 = (blockIdx.x % nbn) * 128;
  v8f acc[4][2];
#pragma unroll
  for (int mt = 0; mt < 4; mt++)
#pragma unroll
    for (int nt = 0; nt < 2; nt++) acc[mt][nt] = v8f_zero();

  auto load_tiles = [&](int buf, int ks) {
    int k0 = ks * 32;
    for (int c = tid; c < 512; c += 256) {
      int row = c >> 2, col8 = (c & 3) * 8;
      async_ld128(&As[buf][row][col8], A + (size_t)(m0 + row) * lda + k0 + col8);
    }
    for (int c = tid; c < 512; c += 256) {
      int row = c >> 2, col8 = (c & 3) * 8;
      async_ld128(&Bs[buf][row][col8], Bt + (size_t)(n0 + row) * ldb + k0 + col8);
    }
  };

  int nk = Kd / 32, buf = 0;
  load_tiles(0, 0);
  wait_async();
  for (int ks = 0; ks < nk; ks++) {
    __syncthreads();
    if (ks + 1 < nk) load_tiles(buf ^ 1, ks + 1);
    BFrag af[4], bf2[2];
#pragma unroll
    for (int mt = 0; mt < 4; mt++) load_fragA(af[mt], &As[buf][wm * 64 + mt * 16 + ln][0], hi);
#pragma unroll
    for (int nt = 0; nt < 2; nt++) load_fragB(bf2[nt], &Bs[buf][wn * 32 + nt * 16 + ln][0], hi);
#pragma unroll
    for (int mt = 0; mt < 4; mt++)
#pragma unroll
      for (int nt = 0; nt < 2; nt++) acc[mt][nt] = wmma_bf16(af[mt], bf2[nt], acc[mt][nt]);
    wait_async();     // next tile landed before the barrier at loop top
    buf ^= 1;
  }

#pragma unroll
  for (int mt = 0; mt < 4; mt++)
#pragma unroll
    for (int nt = 0; nt < 2; nt++)
#pragma unroll
      for (int r = 0; r < 8; r++) {
        int m = m0 + wm * 64 + mt * 16 + r + hi * 8;
        int n = n0 + wn * 32 + nt * 16 + ln;
        float v = acc[mt][nt][r];
        if (BIAS) v += bias[n];
        if (ACC)  v += Cf[(size_t)m * ldc + n];
        if (OUT == 0 || OUT == 2) Cf[(size_t)m * ldc + n] = v;
        if (OUT == 1 || OUT == 2) Cb[(size_t)m * ldc + n] = f32_to_bf16(v);
      }
}

// ---------------- flash attention: 4 waves x 16 queries, 32-key steps ----------------
// NOTE: 1/8 score scale is pre-folded into W_q (wtrans scale), so S is already scaled.
__global__ __launch_bounds__(128) void flash_attn_kernel(
    const unsigned short* __restrict__ qh, const unsigned short* __restrict__ kh,
    const unsigned short* __restrict__ vt, const int* __restrict__ key_mask,
    const int* __restrict__ query_mask, unsigned short* __restrict__ attnvec) {
  __shared__ unsigned short Ks[32][72];      // [key][dh]
  __shared__ unsigned short Vs[64][40];      // [dh][key] (V^T)
  __shared__ unsigned short Ps[4][16][40];   // per-wave P restage

  int tid = threadIdx.x, lane = tid & 31, w = tid >> 5;
  int hi = lane >> 4, ln = lane & 15;
  int blk = blockIdx.x;
  int qt = blk & 31;             // 2048/64 query tiles
  int h  = (blk >> 5) & 7;
  int b  = blk >> 8;
  int q0 = qt * 64 + w * 16;

  BFrag aq[2];
  {
    const unsigned short* base = qh + (size_t)(b * QLEN + q0 + ln) * DMODEL + h * DH;
    load_fragA(aq[0], base, hi);        // d 0..31
    load_fragA(aq[1], base + 32, hi);   // d 32..63
  }

  float rm[8], rl[8];                  // rl = per-lane PARTIAL row sum (deferred reduce)
  v8f o[4];
#pragma unroll
  for (int r = 0; r < 8; r++) { rm[r] = -3.0e38f; rl[r] = 0.f; }
#pragma unroll
  for (int n = 0; n < 4; n++) o[n] = v8f_zero();

  const unsigned short* khb = kh + (size_t)b * KPAD * DMODEL + h * DH;
  const unsigned short* vtb = vt + (size_t)(b * HEADS + h) * DH * KPAD;
  const int* kmb = key_mask + (size_t)b * KLEN;

  for (int j0 = 0; j0 < KPAD; j0 += 32) {
    for (int c = tid; c < 256; c += 128) {      // K tile: 32 x 64 (async -> LDS)
      int row = c >> 3, col = (c & 7) * 8;
      async_ld128(&Ks[row][col], khb + (size_t)(j0 + row) * DMODEL + col);
    }
    for (int c = tid; c < 256; c += 128) {      // V^T tile: 64 x 32 (async -> LDS)
      int row = c >> 2, col = (c & 3) * 8;
      async_ld128(&Vs[row][col], vtb + (size_t)row * KPAD + j0 + col);
    }
    wait_async();
    __syncthreads();

    // S = Q K^T  (K-dim = 64 -> two WMMAs per 16-key column tile)
    v8f s[2];
#pragma unroll
    for (int nt = 0; nt < 2; nt++) {
      v8f z = v8f_zero();
      BFrag bk0, bk1;
      load_fragB(bk0, &Ks[nt * 16 + ln][0], hi);
      load_fragB(bk1, &Ks[nt * 16 + ln][32], hi);
      z = wmma_bf16(aq[0], bk0, z);
      z = wmma_bf16(aq[1], bk1, z);
      s[nt] = z;
    }

    // key mask (scale already folded into W_q)
#pragma unroll
    for (int nt = 0; nt < 2; nt++) {
      int j = j0 + nt * 16 + ln;
      bool valid = (j < KLEN) ? (kmb[j] != 0) : (j == KLEN);
      if (!valid) {
#pragma unroll
        for (int r = 0; r < 8; r++) s[nt][r] = NEG_INF;
      }
    }

    // online softmax: only the row-max needs a cross-lane butterfly per step;
    // the row-sum stays a per-lane partial (corr is uniform across the group).
#pragma unroll
    for (int r = 0; r < 8; r++) {
      float m = fmaxf(s[0][r], s[1][r]);
#pragma unroll
      for (int msk = 1; msk < 16; msk <<= 1) m = fmaxf(m, __shfl_xor(m, msk, 32));
      float nm = fmaxf(rm[r], m);
      float corr = __expf(rm[r] - nm);
      float p0 = __expf(s[0][r] - nm);
      float p1 = __expf(s[1][r] - nm);
      rl[r] = rl[r] * corr + (p0 + p1);
      rm[r] = nm;
      s[0][r] = p0; s[1][r] = p1;
#pragma unroll
      for (int n = 0; n < 4; n++) o[n][r] *= corr;
    }

    // restage P (C layout -> A layout) through per-wave LDS
#pragma unroll
    for (int nt = 0; nt < 2; nt++)
#pragma unroll
      for (int r = 0; r < 8; r++)
        Ps[w][r + hi * 8][nt * 16 + ln] = f32_to_bf16(s[nt][r]);
    __asm__ volatile("" ::: "memory");   // keep DS store->load order (HW is in-order per wave)

    BFrag pf;
    load_fragA(pf, &Ps[w][ln][0], hi);
#pragma unroll
    for (int n = 0; n < 4; n++) {
      BFrag bv;
      load_fragB(bv, &Vs[n * 16 + ln][0], hi);
      o[n] = wmma_bf16(pf, bv, o[n]);
    }
    __syncthreads();
  }

  // finalize deferred row sums (one butterfly per row, once)
#pragma unroll
  for (int r = 0; r < 8; r++)
#pragma unroll
    for (int msk = 1; msk < 16; msk <<= 1) rl[r] += __shfl_xor(rl[r], msk, 32);

  // normalize, apply query mask, store bf16 attn_vec
  float invl[8];
#pragma unroll
  for (int r = 0; r < 8; r++) {
    int q = q0 + r + hi * 8;
    int qm = query_mask[(size_t)b * QLEN + q];
    invl[r] = (qm != 0 ? 1.f : 0.f) / rl[r];
  }
#pragma unroll
  for (int n = 0; n < 4; n++)
#pragma unroll
    for (int r = 0; r < 8; r++) {
      int q = q0 + r + hi * 8;
      int d = h * DH + n * 16 + ln;
      attnvec[(size_t)(b * QLEN + q) * DMODEL + d] = f32_to_bf16(o[n][r] * invl[r]);
    }
}

// ---------------- gating epilogue: out = q + g*q + (1-g)*aproj ----------------
__global__ __launch_bounds__(256) void epilogue_kernel(
    const float* __restrict__ query, const float* __restrict__ glogit,
    const float* __restrict__ aproj, float* __restrict__ out) {
  size_t i = (size_t)blockIdx.x * 256 + threadIdx.x;
  float q = query[i], ap = aproj[i];
  float g = 1.f / (1.f + __expf(-glogit[i]));
  out[i] = q + g * q + (1.f - g) * ap;
}

extern "C" void kernel_launch(void* const* d_in, const int* in_sizes, int n_in,
                              void* d_out, int out_size, void* d_ws, size_t ws_size,
                              hipStream_t stream) {
  (void)in_sizes; (void)n_in; (void)out_size; (void)ws_size;
  const float* query  = (const float*)d_in[0];
  const float* key    = (const float*)d_in[1];
  const float* value  = (const float*)d_in[2];
  const float* wq     = (const float*)d_in[3];
  const float* wk     = (const float*)d_in[4];
  const float* wv     = (const float*)d_in[5];
  const float* wo     = (const float*)d_in[6];
  const float* gw     = (const float*)d_in[7];
  const float* gb     = (const float*)d_in[8];
  const float* qgamma = (const float*)d_in[9];
  const float* qbeta  = (const float*)d_in[10];
  const float* kgamma = (const float*)d_in[11];
  const float* kbeta  = (const float*)d_in[12];
  const float* vgamma = (const float*)d_in[13];
  const float* vbeta  = (const float*)d_in[14];
  const int* qmask    = (const int*)d_in[15];
  const int* kmask    = (const int*)d_in[16];
  float* out = (float*)d_out;

  char* ws = (char*)d_ws;
  size_t off = 0;
  auto alloc = [&](size_t bytes) -> void* {
    off = (off + 255) & ~(size_t)255;
    void* p = ws + off; off += bytes; return p;
  };
  const size_t MQ = (size_t)BATCH * QLEN;   // 4096
  const size_t MK = (size_t)BATCH * KPAD;   // 4224

  unsigned short* qn     = (unsigned short*)alloc(MQ * DMODEL * 2);
  unsigned short* kn     = (unsigned short*)alloc(MK * DMODEL * 2);
  unsigned short* vn     = (unsigned short*)alloc(MK * DMODEL * 2);
  unsigned short* qraw   = (unsigned short*)alloc(MQ * DMODEL * 2);
  unsigned short* wqt    = (unsigned short*)alloc((size_t)DMODEL * DMODEL * 2);
  unsigned short* wkt    = (unsigned short*)alloc((size_t)DMODEL * DMODEL * 2);
  unsigned short* wvt    = (unsigned short*)alloc((size_t)DMODEL * DMODEL * 2);
  unsigned short* wot    = (unsigned short*)alloc((size_t)DMODEL * DMODEL * 2);
  unsigned short* gwt    = (unsigned short*)alloc((size_t)DMODEL * 2 * DMODEL * 2);
  unsigned short* qh     = (unsigned short*)alloc(MQ * DMODEL * 2);
  unsigned short* khp    = (unsigned short*)alloc(MK * DMODEL * 2);
  unsigned short* vhp    = (unsigned short*)alloc(MK * DMODEL * 2);
  unsigned short* vtb    = (unsigned short*)alloc((size_t)BATCH * HEADS * DH * KPAD * 2);
  unsigned short* avec   = (unsigned short*)alloc(MQ * DMODEL * 2);
  float*          aprojf = (float*)alloc(MQ * DMODEL * 4);
  unsigned short* aprojb = (unsigned short*)alloc(MQ * DMODEL * 2);
  float*          glog   = (float*)alloc(MQ * DMODEL * 4);

  // 1) LayerNorms (+ raw query bf16 for the gating matmul)
  ln_q_kernel<<<MQ / 8, 256, 0, stream>>>(query, qgamma, qbeta, qn, qraw);
  ln_kv_kernel<<<MK / 8, 256, 0, stream>>>(key, kgamma, kbeta, kn);
  ln_kv_kernel<<<MK / 8, 256, 0, stream>>>(value, vgamma, vbeta, vn);

  // 2) weights -> bf16, transposed (1/8 attention scale folded into W_q)
  wtrans_kernel<<<(DMODEL / 32) * (DMODEL / 32), 256, 0, stream>>>(wq, wqt, DMODEL, DMODEL, ATT_SCALE);
  wtrans_kernel<<<(DMODEL / 32) * (DMODEL / 32), 256, 0, stream>>>(wk, wkt, DMODEL, DMODEL, 1.f);
  wtrans_kernel<<<(DMODEL / 32) * (DMODEL / 32), 256, 0, stream>>>(wv, wvt, DMODEL, DMODEL, 1.f);
  wtrans_kernel<<<(DMODEL / 32) * (DMODEL / 32), 256, 0, stream>>>(wo, wot, DMODEL, DMODEL, 1.f);
  wtrans_kernel<<<(2 * DMODEL / 32) * (DMODEL / 32), 256, 0, stream>>>(gw, gwt, 2 * DMODEL, DMODEL, 1.f);

  // 3) projections (bf16 WMMA GEMMs)
  gemm_bf16_kernel<1, false, false><<<(MQ / 128) * (DMODEL / 128), 256, 0, stream>>>(
      qn, DMODEL, wqt, DMODEL, nullptr, qh, DMODEL, (int)MQ, DMODEL, DMODEL, nullptr);
  gemm_bf16_kernel<1, false, false><<<(MK / 128) * (DMODEL / 128), 256, 0, stream>>>(
      kn, DMODEL, wkt, DMODEL, nullptr, khp, DMODEL, (int)MK, DMODEL, DMODEL, nullptr);
  gemm_bf16_kernel<1, false, false><<<(MK / 128) * (DMODEL / 128), 256, 0, stream>>>(
      vn, DMODEL, wvt, DMODEL, nullptr, vhp, DMODEL, (int)MK, DMODEL, DMODEL, nullptr);

  // 4) V^T per head for PV WMMA B-operand
  vtrans_kernel<<<BATCH * HEADS * (KPAD / 64), 256, 0, stream>>>(vhp, vtb);

  // 5) attention
  flash_attn_kernel<<<BATCH * HEADS * (QLEN / 64), 128, 0, stream>>>(
      qh, khp, vtb, kmask, qmask, avec);

  // 6) output projection (f32 + bf16 copies)
  gemm_bf16_kernel<2, false, false><<<(MQ / 128) * (DMODEL / 128), 256, 0, stream>>>(
      avec, DMODEL, wot, DMODEL, aprojf, aprojb, DMODEL, (int)MQ, DMODEL, DMODEL, nullptr);

  // 7) gating logits: glog = aprojb @ gwt[:,512:] + g_b ; then += qraw @ gwt[:,:512]
  gemm_bf16_kernel<0, false, true><<<(MQ / 128) * (DMODEL / 128), 256, 0, stream>>>(
      aprojb, DMODEL, gwt + DMODEL, 2 * DMODEL, glog, nullptr, DMODEL, (int)MQ, DMODEL, DMODEL, gb);
  gemm_bf16_kernel<0, true, false><<<(MQ / 128) * (DMODEL / 128), 256, 0, stream>>>(
      qraw, DMODEL, gwt, 2 * DMODEL, glog, nullptr, DMODEL, (int)MQ, DMODEL, DMODEL, nullptr);

  // 8) sigmoid gate + residual
  epilogue_kernel<<<(MQ * DMODEL) / 256, 256, 0, stream>>>(query, glog, aprojf, out);
}
